// FoundationStereoModel_24704651887437
// MI455X (gfx1250) — matrix operand order repaired and checked by reading, not compiled, not run on hardware
//
#include <hip/hip_runtime.h>
#include <cmath>

// ---------------------------------------------------------------------------
// Types for CDNA5 WMMA
// ---------------------------------------------------------------------------
typedef __attribute__((ext_vector_type(16))) __bf16 v16bf;
typedef __attribute__((ext_vector_type(2)))  __bf16 v2bf;
typedef __attribute__((ext_vector_type(8)))  float  v8f;

union FragBF {
    unsigned short u[16];
    unsigned int   w[8];
    v16bf v;
};

// scalar round-to-nearest-even f32 -> bf16 (bit pattern) fallback
__device__ __forceinline__ unsigned short f2bf(float f) {
    unsigned int u = __float_as_uint(f);
    unsigned int r = u + 0x7fffu + ((u >> 16) & 1u);
    return (unsigned short)(r >> 16);
}

// pack two f32 into one dword of bf16 (v_cvt_pk_bf16_f32 when available)
__device__ __forceinline__ unsigned int pack_bf16(float lo, float hi) {
#if __has_builtin(__builtin_amdgcn_cvt_pk_bf16_f32)
    union { v2bf v; unsigned int u; } c;
    c.v = __builtin_amdgcn_cvt_pk_bf16_f32(lo, hi);
    return c.u;
#else
    return (unsigned int)f2bf(lo) | ((unsigned int)f2bf(hi) << 16);
#endif
}

// Async global->LDS copy (CDNA5, ASYNCcnt-tracked). 16 bytes per active lane.
__device__ __forceinline__ void async_load_b128(unsigned lds_off, const float* gaddr) {
    asm volatile("global_load_async_to_lds_b128 %0, %1, off"
                 :: "v"(lds_off), "v"(gaddr)
                 : "memory");
}
__device__ __forceinline__ void wait_asynccnt0() {
    asm volatile("s_wait_asynccnt 0x0" ::: "memory");
}

// ---------------------------------------------------------------------------
// Implicit-GEMM conv2d:  Y[Cout, Hout*Wout] = W[Cout, Cin*K*K] * im2col(X) + b
// Tile: 128(M) x 64(N), BK=32. 256 threads = 8 wave32, each wave owns 32x32
// (2x2 WMMA 16x16 tiles). Double-buffered K-loop:
//   A tile (weights): f32, GLOBAL_LOAD_ASYNC_TO_LDS_B128 DMA for step k+1
//     issued before the WMMAs of step k (DMA hides under matrix work);
//     f32->bf16 at fragment build via v_cvt_pk_bf16_f32.
//   B tile (im2col): VALU gather + packed bf16 ds_store (overlaps the DMA).
// f32 accumulate via v_wmma_f32_16x16x32_bf16.
// ---------------------------------------------------------------------------
template<int KS, int STRIDE>
__global__ __launch_bounds__(256) void conv_gemm_wmma(
    const float* __restrict__ X,    // [Cin, Hin*Win]
    const float* __restrict__ Wt,   // [Cout, Cin*KS*KS] (row-major == OIHW flat)
    const float* __restrict__ Bias, // [Cout]
    float* __restrict__ Y,          // [Cout, Hout*Wout]
    int Cin, int Cout, int Hin, int Win, int Hout, int Wout)
{
    const int Ktot = Cin * KS * KS;
    const int Nout = Hout * Wout;
    const int mBlk = blockIdx.y * 128;
    const int nBlk = blockIdx.x * 64;

    __shared__ __align__(16) float          Af32[2][128 * 32]; // 2 x 16 KB DMA targets
    __shared__ __align__(16) unsigned short Bs[2][32 * 64];    // 2 x 4 KB bf16

    const int t    = threadIdx.x;
    const int lane = t & 31;
    const int wave = t >> 5;
    const int wm   = wave & 3;   // 0..3 : 32-row group
    const int wn   = wave >> 2;  // 0..1 : 32-col group
    const int pad  = KS / 2;
    const int kSteps = Ktot >> 5;   // all layer K dims are multiples of 32

    // Zero-fill A rows beyond Cout once (both buffers); never async-overwritten.
#pragma unroll
    for (int i = 0; i < 4; ++i) {
        const int cid = t + i * 256;          // 0..1023 : 16B chunks of A tile
        const int row = cid >> 3;
        const int col = (cid & 7) << 2;
        if (mBlk + row >= Cout) {
            float4 z = make_float4(0.f, 0.f, 0.f, 0.f);
            *(float4*)&Af32[0][row * 32 + col] = z;
            *(float4*)&Af32[1][row * 32 + col] = z;
        }
    }

    // ---- staging helpers ----
    auto stageA = [&](int ks, int buf) {   // async DMA: 128 rows x 32 f32
        const int k0 = ks << 5;
#pragma unroll
        for (int i = 0; i < 4; ++i) {
            const int cid = t + i * 256;
            const int row = cid >> 3;        // 0..127
            const int col = (cid & 7) << 2;  // 0,4,...,28
            const int m   = mBlk + row;
            if (m < Cout) {
                const float* g = Wt + (size_t)m * Ktot + k0 + col;
                const unsigned lds =
                    (unsigned)(unsigned long long)(const void*)&Af32[buf][row * 32 + col];
                async_load_b128(lds, g);
            }
        }
    };
    auto stageB = [&](int ks, int buf) {   // im2col gather, packed bf16 stores
        const int k0 = ks << 5;
        const int kk = t >> 3;             // 0..31
        const int n0 = (t & 7) << 3;       // 0..56
        const int kg = k0 + kk;
        const int cin = kg / (KS * KS);
        const int rem = kg - cin * (KS * KS);
        const int ky = rem / KS;
        const int kx = rem - ky * KS;
        const float* base = X + (size_t)cin * (Hin * Win);
        unsigned int* dst = (unsigned int*)&Bs[buf][kk * 64 + n0];
#pragma unroll
        for (int i = 0; i < 4; ++i) {
            float v[2];
#pragma unroll
            for (int j = 0; j < 2; ++j) {
                const int n = nBlk + n0 + 2 * i + j;
                float vv = 0.f;
                if (n < Nout) {
                    const int oy = n / Wout;
                    const int ox = n - oy * Wout;
                    const int iy = oy * STRIDE + ky - pad;
                    const int ix = ox * STRIDE + kx - pad;
                    if ((unsigned)iy < (unsigned)Hin && (unsigned)ix < (unsigned)Win)
                        vv = base[iy * Win + ix];
                }
                v[j] = vv;
            }
            dst[i] = pack_bf16(v[0], v[1]);
        }
    };

    v8f acc[2][2] = {};

    // prologue: fill buffer 0
    stageA(0, 0);
    stageB(0, 0);
    wait_asynccnt0();
    __syncthreads();

    for (int ks = 0; ks < kSteps; ++ks) {
        const int buf = ks & 1;

        // issue next tile's DMA + staging before this tile's WMMAs
        if (ks + 1 < kSteps) {
            stageA(ks + 1, buf ^ 1);
            stageB(ks + 1, buf ^ 1);
        }

        // ---- fragment loads per ISA VGPR layout ----
        // A 16x32 bf16: lanes 0-15 -> M=0..15 K=[0..7,16..23]; lanes16-31 K=[8..15,24..31]
        FragBF a[2], b[2];
        const int mrow  = (wm << 5) + (lane & 15);
        const int kbase = (lane >> 4) << 3;
#pragma unroll
        for (int mt = 0; mt < 2; ++mt) {
            const float* p = &Af32[buf][(mrow + mt * 16) * 32 + kbase];
#pragma unroll
            for (int i = 0; i < 4; ++i) {
                a[mt].w[i]     = pack_bf16(p[2 * i],      p[2 * i + 1]);
                a[mt].w[4 + i] = pack_bf16(p[16 + 2 * i], p[17 + 2 * i]);
            }
        }
        // B 32x16 bf16: lanes 0-15 -> N=0..15 K=[0..7,16..23]; lanes16-31 K=[8..15,24..31]
#pragma unroll
        for (int nt = 0; nt < 2; ++nt) {
            const unsigned short* p = &Bs[buf][(wn << 5) + nt * 16 + (lane & 15)];
#pragma unroll
            for (int i = 0; i < 8; ++i) { b[nt].u[i] = p[(kbase + i) * 64]; b[nt].u[8 + i] = p[(16 + kbase + i) * 64]; }
        }

#pragma unroll
        for (int mt = 0; mt < 2; ++mt)
#pragma unroll
            for (int nt = 0; nt < 2; ++nt)
                acc[mt][nt] = __builtin_amdgcn_wmma_f32_16x16x32_bf16(
                    false, a[mt].v, false, b[nt].v, (short)0, acc[mt][nt], false, false);

        wait_asynccnt0();      // next-buffer DMA complete (no-op on last step)
        __syncthreads();       // all waves' staging visible / compute drained
    }

    __builtin_amdgcn_s_wait_tensorcnt(0);

    // ---- epilogue: C/D layout: VGPR j -> M = j + 8*(lane/16), N = lane%16 ----
#pragma unroll
    for (int mt = 0; mt < 2; ++mt) {
#pragma unroll
        for (int nt = 0; nt < 2; ++nt) {
            const int n = nBlk + (wn << 5) + nt * 16 + (lane & 15);
            if (n >= Nout) continue;
            const int mb = mBlk + (wm << 5) + mt * 16 + ((lane >> 4) << 3);
#pragma unroll
            for (int j = 0; j < 8; ++j) {
                const int m = mb + j;
                if (m < Cout) Y[(size_t)m * Nout + n] = acc[mt][nt][j] + Bias[m];
            }
        }
    }
}

// ---------------------------------------------------------------------------
// GroupNorm (32 groups) with fused activation.
// mode 1: silu(gn(h));   mode 2: silu(x + gn(h))   (resblock tail)
// ---------------------------------------------------------------------------
__global__ __launch_bounds__(256) void groupnorm_kernel(
    const float* __restrict__ h, const float* __restrict__ xres,
    float* __restrict__ out,
    const float* __restrict__ gamma, const float* __restrict__ beta,
    int C, int HW, int mode)
{
    const int grp = blockIdx.x;       // 0..31
    const int cpg = C / 32;
    const size_t base = (size_t)grp * cpg * HW;
    const int cnt = cpg * HW;

    float s = 0.f, ss = 0.f;
    for (int i = threadIdx.x; i < cnt; i += 256) {
        const float v = h[base + i];
        s += v; ss += v * v;
    }
    __shared__ float rs[256], rq[256];
    rs[threadIdx.x] = s; rq[threadIdx.x] = ss;
    __syncthreads();
    for (int off = 128; off > 0; off >>= 1) {
        if (threadIdx.x < off) { rs[threadIdx.x] += rs[threadIdx.x + off]; rq[threadIdx.x] += rq[threadIdx.x + off]; }
        __syncthreads();
    }
    const float inv = 1.f / (float)cnt;
    const float mu  = rs[0] * inv;
    const float var = rq[0] * inv - mu * mu;
    const float rstd = rsqrtf(var + 1e-5f);

    for (int i = threadIdx.x; i < cnt; i += 256) {
        const int c = grp * cpg + i / HW;
        float v = (h[base + i] - mu) * rstd * gamma[c] + beta[c];
        if (mode == 2) v += xres[base + i];
        v = v / (1.f + __expf(-v));   // SiLU
        out[base + i] = v;
    }
}

// ---------------------------------------------------------------------------
// Pixel shuffle r=2: in [Cin,H,W] -> out [Cin/4, 2H, 2W]
// ---------------------------------------------------------------------------
__global__ void pixel_shuffle_kernel(const float* __restrict__ in, float* __restrict__ out,
                                     int Cin, int H, int W)
{
    const int idx = blockIdx.x * 256 + threadIdx.x;
    const int total = Cin * H * W;
    if (idx >= total) return;
    const int x = idx % W;
    int tmp = idx / W;
    const int y = tmp % H;
    const int ci = tmp / H;
    const int co = ci >> 2, r = ci & 3, ry = r >> 1, rx = r & 1;
    const int Wo = W * 2;
    out[((size_t)co * (H * 2) + (y * 2 + ry)) * Wo + (x * 2 + rx)] = in[idx];
}

// ---------------------------------------------------------------------------
// Cost volume: group-wise correlation + concat(proj) + corr_stem 1x1x1.
// comb[(o*104+d)*2304 + hw]   (matches reshape(CV*D, H, W))
// ---------------------------------------------------------------------------
__global__ __launch_bounds__(256) void cost_volume_kernel(
    const float* __restrict__ fl, const float* __restrict__ fr,
    const float* __restrict__ lp, const float* __restrict__ rp,
    const float* __restrict__ stem_w, const float* __restrict__ stem_b,
    float* __restrict__ comb)
{
    const int idx = blockIdx.x * 256 + threadIdx.x;   // over D*H*W
    if (idx >= 104 * 48 * 48) return;
    const int x = idx % 48;
    const int t2 = idx / 48;
    const int y = t2 % 48;
    const int d = t2 / 48;
    const int hw = y * 48 + x;
    const int xm = x - d;
    const bool valid = (xm >= 0);
    const int hwm = y * 48 + (valid ? xm : 0);

    float cin[32];
#pragma unroll
    for (int g = 0; g < 8; ++g) {
        float acc = 0.f;
        if (valid) {
            for (int c = 0; c < 16; ++c) {
                const int ch = g * 16 + c;
                acc += fl[ch * 2304 + hw] * fr[ch * 2304 + hwm];
            }
            acc *= (1.f / 16.f);
        }
        cin[g] = acc;
    }
#pragma unroll
    for (int c = 0; c < 12; ++c) cin[8 + c]  = valid ? lp[c * 2304 + hw]  : 0.f;
#pragma unroll
    for (int c = 0; c < 12; ++c) cin[20 + c] = valid ? rp[c * 2304 + hwm] : 0.f;

    for (int o = 0; o < 28; ++o) {
        float a = stem_b[o];
#pragma unroll
        for (int c = 0; c < 32; ++c) a += stem_w[o * 32 + c] * cin[c];
        comb[((size_t)(o * 104 + d)) * 2304 + hw] = a;
    }
}

// ---------------------------------------------------------------------------
// Classifier: logits = cls_w . (comb + y), softmax over D, disparity -> depth
// ---------------------------------------------------------------------------
__global__ __launch_bounds__(128) void classifier_kernel(
    const float* __restrict__ comb, const float* __restrict__ y,
    const float* __restrict__ cls_w, const float* __restrict__ cls_b,
    float* __restrict__ out)
{
    const int hw = blockIdx.x;       // 0..2303
    const int d  = threadIdx.x;      // 0..127
    __shared__ float sl[128];
    __shared__ float red[128];

    float logit = -1e30f;
    if (d < 104) {
        float a = cls_b[0];
        for (int c = 0; c < 28; ++c) {
            const size_t off = ((size_t)(c * 104 + d)) * 2304 + hw;
            a += cls_w[c] * (comb[off] + y[off]);
        }
        logit = a;
    }
    sl[d] = logit;
    red[d] = logit;
    __syncthreads();
    for (int off = 64; off > 0; off >>= 1) {
        if (d < off) red[d] = fmaxf(red[d], red[d + off]);
        __syncthreads();
    }
    const float mx = red[0];
    __syncthreads();
    const float e = (d < 104) ? __expf(sl[d] - mx) : 0.f;
    red[d] = e;
    __syncthreads();
    for (int off = 64; off > 0; off >>= 1) {
        if (d < off) red[d] += red[d + off];
        __syncthreads();
    }
    const float denom = red[0];
    __syncthreads();
    red[d] = e * (float)d;
    __syncthreads();
    for (int off = 64; off > 0; off >>= 1) {
        if (d < off) red[d] += red[d + off];
        __syncthreads();
    }
    if (d == 0) {
        const float disp = red[0] / denom;
        const int xc = hw % 48;
        const float depth = (721.5f * 0.54f) / fmaxf(disp, 1e-6f);
        out[hw] = (((float)xc - disp) < 0.f) ? __builtin_inff() : depth;
    }
}

// ---------------------------------------------------------------------------
// Host-side helpers
// ---------------------------------------------------------------------------
static inline void launch_conv(hipStream_t s,
                               const float* X, const float* W, const float* B, float* Y,
                               int Cin, int Cout, int Hin, int Win, int ks, int stride)
{
    const int Hout = (Hin + 2 * (ks / 2) - ks) / stride + 1;
    const int Wout = (Win + 2 * (ks / 2) - ks) / stride + 1;
    const int Nout = Hout * Wout;
    dim3 grid((Nout + 63) / 64, (Cout + 127) / 128);
    dim3 blk(256);
    if (ks == 1)
        conv_gemm_wmma<1, 1><<<grid, blk, 0, s>>>(X, W, B, Y, Cin, Cout, Hin, Win, Hout, Wout);
    else if (stride == 1)
        conv_gemm_wmma<3, 1><<<grid, blk, 0, s>>>(X, W, B, Y, Cin, Cout, Hin, Win, Hout, Wout);
    else
        conv_gemm_wmma<3, 2><<<grid, blk, 0, s>>>(X, W, B, Y, Cin, Cout, Hin, Win, Hout, Wout);
}

static inline void launch_gn(hipStream_t s, const float* h, const float* x, float* out,
                             const float* g, const float* b, int C, int HW, int mode)
{
    groupnorm_kernel<<<32, 256, 0, s>>>(h, x, out, g, b, C, HW, mode);
}

extern "C" void kernel_launch(void* const* d_in, const int* in_sizes, int n_in,
                              void* d_out, int out_size, void* d_ws, size_t ws_size,
                              hipStream_t stream)
{
    // ---- inputs (setup_inputs order) ----
    const float* fl     = (const float*)d_in[0];
    const float* fr     = (const float*)d_in[1];
    const float* proj_w = (const float*)d_in[2];
    const float* proj_b = (const float*)d_in[3];
    const float* stem_w = (const float*)d_in[4];
    const float* stem_b = (const float*)d_in[5];
    const float* c0_w = (const float*)d_in[6];   const float* c0_b = (const float*)d_in[7];
    const float* gn0_g = (const float*)d_in[8];  const float* gn0_b = (const float*)d_in[9];
    const float* c1_w = (const float*)d_in[10];  const float* c1_b = (const float*)d_in[11];
    const float* gn1_g = (const float*)d_in[12]; const float* gn1_b = (const float*)d_in[13];
    const float* r1a_w = (const float*)d_in[14]; const float* r1a_b = (const float*)d_in[15];
    const float* r1gna_g = (const float*)d_in[16]; const float* r1gna_b = (const float*)d_in[17];
    const float* r1b_w = (const float*)d_in[18]; const float* r1b_b = (const float*)d_in[19];
    const float* r1gnb_g = (const float*)d_in[20]; const float* r1gnb_b = (const float*)d_in[21];
    const float* c2_w = (const float*)d_in[22];  const float* c2_b = (const float*)d_in[23];
    const float* gn2_g = (const float*)d_in[24]; const float* gn2_b = (const float*)d_in[25];
    const float* r2a_w = (const float*)d_in[26]; const float* r2a_b = (const float*)d_in[27];
    const float* r2gna_g = (const float*)d_in[28]; const float* r2gna_b = (const float*)d_in[29];
    const float* r2b_w = (const float*)d_in[30]; const float* r2b_b = (const float*)d_in[31];
    const float* r2gnb_g = (const float*)d_in[32]; const float* r2gnb_b = (const float*)d_in[33];
    const float* c3_w = (const float*)d_in[34];  const float* c3_b = (const float*)d_in[35];
    const float* gn3_g = (const float*)d_in[36]; const float* gn3_b = (const float*)d_in[37];
    const float* d0_w = (const float*)d_in[38];  const float* d0_b = (const float*)d_in[39];
    const float* dgn0_g = (const float*)d_in[40]; const float* dgn0_b = (const float*)d_in[41];
    const float* d1_w = (const float*)d_in[42];  const float* d1_b = (const float*)d_in[43];
    const float* dgn1_g = (const float*)d_in[44]; const float* dgn1_b = (const float*)d_in[45];
    const float* d2_w = (const float*)d_in[46];  const float* d2_b = (const float*)d_in[47];
    const float* cls_w = (const float*)d_in[48]; const float* cls_b = (const float*)d_in[49];

    float* out = (float*)d_out;   // [48*48] depth

    // ---- workspace carve (floats) ----
    float* ws = (float*)d_ws;
    size_t o = 0;
    float* lp     = ws + o; o += 12 * 2304;
    float* rp     = ws + o; o += 12 * 2304;
    float* comb   = ws + o; o += (size_t)2912 * 2304;   // 6,709,248
    float* big1   = ws + o; o += 7000000;
    float* big2   = ws + o; o += 7000000;
    float* small3 = ws + o; o += 1500000;
    if (ws_size < o * sizeof(float)) return;   // insufficient scratch

    // ---- proj (1x1 conv 128 -> 12) on both feature maps ----
    launch_conv(stream, fl, proj_w, proj_b, lp, 128, 12, 48, 48, 1, 1);
    launch_conv(stream, fr, proj_w, proj_b, rp, 128, 12, 48, 48, 1, 1);

    // ---- cost volume + corr stem -> comb [2912, 2304] ----
    {
        const int total = 104 * 48 * 48;
        cost_volume_kernel<<<(total + 255) / 256, 256, 0, stream>>>(
            fl, fr, lp, rp, stem_w, stem_b, comb);
    }

    // ---- compressor ----
    launch_conv(stream, comb, c0_w, c0_b, big1, 2912, 2688, 48, 48, 1, 1);
    launch_gn(stream, big1, nullptr, big1, gn0_g, gn0_b, 2688, 2304, 1);

    launch_conv(stream, big1, c1_w, c1_b, big2, 2688, 2496, 48, 48, 3, 2);   // -> 24x24
    launch_gn(stream, big2, nullptr, big2, gn1_g, gn1_b, 2496, 576, 1);

    // resblock r1 (x = big2)
    launch_conv(stream, big2, r1a_w, r1a_b, small3, 2496, 2496, 24, 24, 3, 1);
    launch_gn(stream, small3, nullptr, small3, r1gna_g, r1gna_b, 2496, 576, 1);
    launch_conv(stream, small3, r1b_w, r1b_b, big1, 2496, 2496, 24, 24, 3, 1);
    launch_gn(stream, big1, big2, big2, r1gnb_g, r1gnb_b, 2496, 576, 2);

    launch_conv(stream, big2, c2_w, c2_b, big1, 2496, 2272, 24, 24, 3, 2);   // -> 12x12
    launch_gn(stream, big1, nullptr, big1, gn2_g, gn2_b, 2272, 144, 1);

    // resblock r2 (x = big1)
    launch_conv(stream, big1, r2a_w, r2a_b, small3, 2272, 2272, 12, 12, 3, 1);
    launch_gn(stream, small3, nullptr, small3, r2gna_g, r2gna_b, 2272, 144, 1);
    launch_conv(stream, small3, r2b_w, r2b_b, big2, 2272, 2272, 12, 12, 3, 1);
    launch_gn(stream, big2, big1, big1, r2gnb_g, r2gnb_b, 2272, 144, 2);

    launch_conv(stream, big1, c3_w, c3_b, big2, 2272, 1024, 12, 12, 3, 2);   // -> 6x6
    launch_gn(stream, big2, nullptr, big2, gn3_g, gn3_b, 1024, 36, 1);

    // ---- decompressor ----
    launch_conv(stream, big2, d0_w, d0_b, big1, 1024, 4096, 6, 6, 3, 1);     // [4096, 36]
    pixel_shuffle_kernel<<<(4096 * 36 + 255) / 256, 256, 0, stream>>>(big1, small3, 4096, 6, 6);
    launch_gn(stream, small3, nullptr, small3, dgn0_g, dgn0_b, 1024, 144, 1);

    launch_conv(stream, small3, d1_w, d1_b, big1, 1024, 2048, 12, 12, 3, 1); // [2048, 144]
    pixel_shuffle_kernel<<<(2048 * 144 + 255) / 256, 256, 0, stream>>>(big1, big2, 2048, 12, 12);
    launch_gn(stream, big2, nullptr, big2, dgn1_g, dgn1_b, 512, 576, 1);

    launch_conv(stream, big2, d2_w, d2_b, big1, 512, 11648, 24, 24, 3, 1);   // [11648, 576]
    pixel_shuffle_kernel<<<(11648 * 576 + 255) / 256, 256, 0, stream>>>(big1, big2, 11648, 24, 24);
    // big2 now [2912, 2304] = y

    // ---- classifier + disparity regression + depth ----
    classifier_kernel<<<2304, 128, 0, stream>>>(comb, big2, cls_w, cls_b, out);
}
